// MultiClippedFourierPSNR_34196529611265
// MI455X (gfx1250) — compile-verified, matrix operand order checked by reading, not run on hardware
//
#include <hip/hip_runtime.h>
#include <hip/hip_bf16.h>

typedef __attribute__((ext_vector_type(16))) _Float16 v16h;
typedef __attribute__((ext_vector_type(8)))  float    v8f;

#define IMS   128
#define PITCH 132   // f16 elements per LDS row (dword-aligned, bank-spread)
#define NT    64

__device__ __forceinline__ v8f wmma16(v16h a, v16h b, v8f c) {
  // D = A(16x32 f16) * B(32x16 f16) + C(16x16 f32)
  return __builtin_amdgcn_wmma_f32_16x16x32_f16(
      /*neg_a=*/false, a, /*neg_b=*/false, b,
      /*c_mod=*/(short)0, c, /*reuse_a=*/false, /*reuse_b=*/false);
}

// Load one WMMA operand fragment from an LDS matrix stored so that the
// per-lane "row" index is contiguous-K. Per CDNA5 ISA 16-bit A (16x32 MxK)
// layout: lane L holds row M=L&15; khalf=L>>4 selects K sub-block; element e
// maps to K = kbase + 16*(e>>3) + 8*khalf + (e&7). The 32x16 B operand uses
// the identical per-lane K mapping with N=L&15, so one helper serves both
// (B-side matrices are stored transposed, i.e. [N][K]).
__device__ __forceinline__ v16h load_frag(const _Float16* m, int row, int kbase, int khalf) {
  const _Float16* p = m + row * PITCH + kbase + khalf * 8;
  v16h v;
#pragma unroll
  for (int j = 0; j < 8; ++j) {
    v[j]     = p[j];        // e = 0..7   -> K = kbase + 8*khalf + j
    v[j + 8] = p[j + 16];   // e = 8..15  -> K = kbase + 16 + 8*khalf + j
  }
  return v;
}

extern "C" __global__ __launch_bounds__(256)
void mcf_psnr_kernel(const float* __restrict__ pred,
                     const float* __restrict__ gt,
                     float* __restrict__ out) {
  extern __shared__ char smem[];
  _Float16* sWr = (_Float16*)smem;            // DFT twiddle real, [k][n]
  _Float16* sWi = sWr + IMS * PITCH;          // DFT twiddle imag, [k][n]
  _Float16* sDt = sWi + IMS * PITCH;          // (pred-gt) transposed, [x][y]
  _Float16* sAr = sDt + IMS * PITCH;          // stage-1 real,  [m][k]
  _Float16* sAi = sAr + IMS * PITCH;          // stage-1 imag,  [m][k]
  float*    bins = (float*)(sAi + IMS * PITCH);  // 65 slots (64 used)
  int*      thr2 = (int*)(bins + 72);            // 64 squared thresholds

  const int tid   = threadIdx.x;
  const int lane  = tid & 31;
  const int wave  = tid >> 5;        // wave32: 8 waves per block
  const int khalf = lane >> 4;
  const int l16   = lane & 15;
  const int b     = blockIdx.x;      // one workgroup per batch element

  // ---- one-time setup: thresholds, bins, DFT twiddles ----
  if (tid < NT) {
    float x = 2.0f + (88.0f / 63.0f) * (float)tid;   // linspace(2, 90, 64)
    float r = rintf(x);                               // half-to-even, matches np.round
    thr2[tid] = (int)(r * r + 0.5f);
  }
  if (tid < 65) bins[tid] = 0.0f;
  for (int idx = tid; idx < IMS * IMS; idx += 256) {
    int k = idx >> 7, n = idx & 127;
    int t = (k * n) & 127;                            // exp(-2*pi*i*k*n/128)
    float ang = (float)t * (-0.04908738521234052f);   // -2*pi/128
    float s, c;
    __sincosf(ang, &s, &c);
    sWr[k * PITCH + n] = (_Float16)c;
    sWi[k * PITCH + n] = (_Float16)s;
  }
  __syncthreads();

  for (int ch = 0; ch < 3; ++ch) {
    const size_t base = ((size_t)b * 3 + ch) * (IMS * IMS);
    __syncthreads();   // prev stage-1 done reading sDt / stage-2 done reading sAr/sAi

    // ---- build D^T = (pred - gt)^T in f16; global reads fully coalesced ----
    for (int idx = tid; idx < IMS * IMS; idx += 256) {
      int y = idx >> 7, x = idx & 127;
      float d = pred[base + idx] - gt[base + idx];
      sDt[x * PITCH + y] = (_Float16)d;
    }
    __syncthreads();

    // ---- stage 1: A = W * D (complex = 2 real GEMMs), 64 tiles / 8 waves ----
    for (int i = 0; i < 8; ++i) {
      int tt = wave + (i << 3);
      int tm = tt >> 3, tn = tt & 7;
      v8f ar = {}, ai = {};
#pragma unroll
      for (int kk = 0; kk < IMS; kk += 32) {
        v16h bf = load_frag(sDt, tn * 16 + l16, kk, khalf);  // B: D (stored [n][k])
        v16h wr = load_frag(sWr, tm * 16 + l16, kk, khalf);  // A: W rows
        v16h wi = load_frag(sWi, tm * 16 + l16, kk, khalf);
        ar = wmma16(wr, bf, ar);
        ai = wmma16(wi, bf, ai);
      }
      // C/D layout: vgpr r -> M = r + 8*khalf, N = lane&15
#pragma unroll
      for (int r = 0; r < 8; ++r) {
        int m = tm * 16 + r + khalf * 8;
        int n = tn * 16 + l16;
        sAr[m * PITCH + n] = (_Float16)ar[r];
        sAi[m * PITCH + n] = (_Float16)ai[r];
      }
    }
    __syncthreads();

    // ---- stage 2: F = A * W^T ; |F|^2 -> radial bins ----
    // B operand = W^T, whose [N][K] storage is W row-major itself.
    for (int i = 0; i < 8; ++i) {
      int tt = wave + (i << 3);
      int tm = tt >> 3, tn = tt & 7;
      v8f p1 = {}, p2 = {}, p3 = {}, p4 = {};
#pragma unroll
      for (int kk = 0; kk < IMS; kk += 32) {
        v16h af = load_frag(sAr, tm * 16 + l16, kk, khalf);
        v16h bf = load_frag(sAi, tm * 16 + l16, kk, khalf);
        v16h wr = load_frag(sWr, tn * 16 + l16, kk, khalf);
        v16h wi = load_frag(sWi, tn * 16 + l16, kk, khalf);
        p1 = wmma16(af, wr, p1);   // Ar*Wr^T
        p2 = wmma16(bf, wi, p2);   // Ai*Wi^T
        p3 = wmma16(af, wi, p3);   // Ar*Wi^T
        p4 = wmma16(bf, wr, p4);   // Ai*Wr^T
      }
#pragma unroll
      for (int r = 0; r < 8; ++r) {
        int m = tm * 16 + r + khalf * 8;   // frequency ky
        int n = tn * 16 + l16;             // frequency kx
        float fr = p1[r] - p2[r];
        float fi = p3[r] + p4[r];
        float sq = fr * fr + fi * fi;
        int dy = (m < 64) ? m : m - 128;   // signed freq == fftshift distance
        int dx = (n < 64) ? n : n - 128;
        int r2 = dx * dx + dy * dy;
        if (r2 <= thr2[NT - 1]) {          // inside largest mask (corners dropped)
          // branchless lower_bound over 64 sorted thr2: first t with thr2[t] >= r2
          int t = 0;
#pragma unroll
          for (int step = 32; step >= 1; step >>= 1)
            if (thr2[t + step - 1] < r2) t += step;
          atomicAdd(&bins[t], sq);         // ds_add_f32
        }
      }
    }
  }
  __syncthreads();

  // ---- prefix over bins (nested masks) + PSNR ----
  if (tid < NT) {
    float s = 0.0f;
    for (int j = 0; j <= tid; ++j) s += bins[j];
    float cst = 20.0f * log10f(16384.0f);  // 20*log10(IM_MAX * H * W)
    out[b * NT + tid] = cst - 10.0f * log10f(s);
  }
}

extern "C" void kernel_launch(void* const* d_in, const int* in_sizes, int n_in,
                              void* d_out, int out_size, void* d_ws, size_t ws_size,
                              hipStream_t stream) {
  const float* pred = (const float*)d_in[0];
  const float* gt   = (const float*)d_in[1];
  float* out = (float*)d_out;

  size_t smem = (size_t)5 * IMS * PITCH * sizeof(_Float16)  // 5 matrices
              + 72 * sizeof(float)                          // bins (+pad)
              + 64 * sizeof(int);                           // thr2
  hipLaunchKernelGGL(mcf_psnr_kernel, dim3(256), dim3(256), smem, stream,
                     pred, gt, out);
}
